// _MoEBlock_2499670966557
// MI455X (gfx1250) — matrix operand order repaired and checked by reading, not compiled, not run on hardware
//
#include <hip/hip_runtime.h>

// Problem shape (fixed by reference): B=4, S=2048, H=2048, E=8
constexpr int Hdim = 2048;
constexpr int Edim = 8;
constexpr int Ntok = 4 * 2048;   // 8192 tokens

typedef __attribute__((ext_vector_type(16))) __bf16 bf16x16;
typedef __attribute__((ext_vector_type(8)))  __bf16 bf16x8;
typedef __attribute__((ext_vector_type(8)))  float  f32x8;

union FragAB { bf16x16 v; bf16x8 h[2]; };
union FragC  { f32x8 v; float f[8]; };
union Pack2  { __bf16 b[2]; unsigned u; };

#if defined(__has_builtin)
#  if __has_builtin(__builtin_amdgcn_global_load_async_to_lds_b128)
#    define HAVE_ASYNC_LDS 1
#  endif
#endif
#ifndef HAVE_ASYNC_LDS
#  define HAVE_ASYNC_LDS 0
#endif

// Exact pointer types for the async builtin: 16-byte int vector in AS1 / AS3.
typedef int v4i_t __attribute__((vector_size(16)));
typedef __attribute__((address_space(1))) v4i_t* g128p;
typedef __attribute__((address_space(3))) v4i_t* l128p;

__device__ __forceinline__ void async_cp16(const void* g, void* l) {
#if HAVE_ASYNC_LDS
  // AS1 value == flat VA; AS3 offset == low 32 bits of the flat LDS pointer
  // (per the ISA's LDS aperture). Use inttoptr casts.
  __builtin_amdgcn_global_load_async_to_lds_b128(
      (g128p)(unsigned long long)(uintptr_t)g,
      (l128p)(unsigned)(uintptr_t)l, 0, 0);
#else
  (void)g; (void)l;
#endif
}

__device__ __forceinline__ void wait_async0() {
#if defined(__has_builtin) && __has_builtin(__builtin_amdgcn_s_wait_asynccnt)
  __builtin_amdgcn_s_wait_asynccnt(0);
#elif defined(__AMDGCN__)
  asm volatile("s_wait_asynccnt 0x0" ::: "memory");
#endif
}

// ---------------------------------------------------------------------------
// 0) zero the per-expert counters (ws is poisoned between runs)
// ---------------------------------------------------------------------------
__global__ __launch_bounds__(32) void moe_init_kernel(int* __restrict__ counts) {
  if (threadIdx.x < Edim) counts[threadIdx.x] = 0;
}

// ---------------------------------------------------------------------------
// 1) gating: one wave32 per token; 8 logits -> softmax top-1 -> eid/gate/count
// ---------------------------------------------------------------------------
__global__ __launch_bounds__(256) void moe_gate_kernel(
    const float* __restrict__ x, const float* __restrict__ wg,
    int* __restrict__ eid, float* __restrict__ gatew, int* __restrict__ counts) {
  const int token = blockIdx.x * 8 + (threadIdx.x >> 5);
  const int lane  = threadIdx.x & 31;
  if (token >= Ntok) return;

  const float* __restrict__ xr = x + (size_t)token * Hdim;
  float acc[Edim];
#pragma unroll
  for (int e = 0; e < Edim; ++e) acc[e] = 0.0f;

  for (int h = lane; h < Hdim; h += 32) {
    const float  xv = xr[h];
    const float4 w0 = *(const float4*)(wg + (size_t)h * Edim);
    const float4 w1 = *(const float4*)(wg + (size_t)h * Edim + 4);
    acc[0] += xv * w0.x; acc[1] += xv * w0.y;
    acc[2] += xv * w0.z; acc[3] += xv * w0.w;
    acc[4] += xv * w1.x; acc[5] += xv * w1.y;
    acc[6] += xv * w1.z; acc[7] += xv * w1.w;
  }

#pragma unroll
  for (int e = 0; e < Edim; ++e) {
#pragma unroll
    for (int m = 16; m >= 1; m >>= 1) acc[e] += __shfl_xor(acc[e], m, 32);
  }

  if (lane == 0) {
    int   best = 0;
    float bl   = acc[0];
#pragma unroll
    for (int e = 1; e < Edim; ++e) {        // strict > keeps first-max (argmax)
      if (acc[e] > bl) { bl = acc[e]; best = e; }
    }
    float s = 0.0f;
#pragma unroll
    for (int e = 0; e < Edim; ++e) s += __expf(acc[e] - bl);
    eid[token]   = best;
    gatew[token] = 1.0f / s;                // softmax value at the max logit
    atomicAdd(&counts[best], 1);
  }
}

// ---------------------------------------------------------------------------
// 2) exclusive scan over 8 counts; reset scatter cursors
// ---------------------------------------------------------------------------
__global__ __launch_bounds__(32) void moe_scan_kernel(
    const int* __restrict__ counts, int* __restrict__ offsets, int* __restrict__ cursor) {
  if (threadIdx.x == 0) {
    int s = 0;
#pragma unroll
    for (int e = 0; e < Edim; ++e) { offsets[e] = s; s += counts[e]; cursor[e] = 0; }
  }
}

// ---------------------------------------------------------------------------
// 3) scatter tokens into per-expert contiguous lists
// ---------------------------------------------------------------------------
__global__ __launch_bounds__(256) void moe_scatter_kernel(
    const int* __restrict__ eid, const int* __restrict__ offsets,
    int* __restrict__ cursor, int* __restrict__ perm) {
  const int t = blockIdx.x * 256 + threadIdx.x;
  if (t < Ntok) {
    const int e = eid[t];
    const int p = atomicAdd(&cursor[e], 1);
    perm[offsets[e] + p] = t;
  }
}

// ---------------------------------------------------------------------------
// 3b) pre-convert x -> bf16 hi/lo planes (row-major [N][K])
// ---------------------------------------------------------------------------
__global__ __launch_bounds__(256) void moe_xconv_kernel(
    const float* __restrict__ x, unsigned short* __restrict__ xh,
    unsigned short* __restrict__ xl) {
  const size_t i4 = ((size_t)blockIdx.x * 256 + threadIdx.x) * 4;
  const float4 v = *(const float4*)(x + i4);
  Pack2 h0, h1, l0, l1;
  h0.b[0] = (__bf16)v.x; h0.b[1] = (__bf16)v.y;
  h1.b[0] = (__bf16)v.z; h1.b[1] = (__bf16)v.w;
  l0.b[0] = (__bf16)(v.x - (float)h0.b[0]); l0.b[1] = (__bf16)(v.y - (float)h0.b[1]);
  l1.b[0] = (__bf16)(v.z - (float)h1.b[0]); l1.b[1] = (__bf16)(v.w - (float)h1.b[1]);
  *(unsigned*)(xh + i4)     = h0.u;
  *(unsigned*)(xh + i4 + 2) = h1.u;
  *(unsigned*)(xl + i4)     = l0.u;
  *(unsigned*)(xl + i4 + 2) = l1.u;
}

// ---------------------------------------------------------------------------
// 3c) pre-convert + transpose W[e][o][k] -> Wt{h,l}[e][k][o] bf16 planes
//     (32x32 tile transpose through LDS, coalesced both directions)
// ---------------------------------------------------------------------------
__global__ __launch_bounds__(256) void moe_wconv_kernel(
    const float* __restrict__ W, unsigned short* __restrict__ wth,
    unsigned short* __restrict__ wtl) {
  __shared__ float T[32][33];
  const int e  = blockIdx.z;
  const int h0 = blockIdx.x * 32;   // k (input) tile
  const int o0 = blockIdx.y * 32;   // o (output) tile
  const int t  = threadIdx.x;

  {
    const int ol = t >> 3;          // 0..31
    const int hl = (t & 7) * 4;     // 0..28
    const float4 v = *(const float4*)(W + ((size_t)e * Hdim + (o0 + ol)) * Hdim + h0 + hl);
    T[hl + 0][ol] = v.x; T[hl + 1][ol] = v.y;
    T[hl + 2][ol] = v.z; T[hl + 3][ol] = v.w;
  }
  __syncthreads();
  {
    const int hl = t >> 3;          // 0..31
    const int ol = (t & 7) * 4;     // 0..28
    const float u0 = T[hl][ol + 0], u1 = T[hl][ol + 1];
    const float u2 = T[hl][ol + 2], u3 = T[hl][ol + 3];
    Pack2 ph0, ph1, pl0, pl1;
    ph0.b[0] = (__bf16)u0; ph0.b[1] = (__bf16)u1;
    ph1.b[0] = (__bf16)u2; ph1.b[1] = (__bf16)u3;
    pl0.b[0] = (__bf16)(u0 - (float)ph0.b[0]); pl0.b[1] = (__bf16)(u1 - (float)ph0.b[1]);
    pl1.b[0] = (__bf16)(u2 - (float)ph1.b[0]); pl1.b[1] = (__bf16)(u3 - (float)ph1.b[1]);
    const size_t base = ((size_t)e * Hdim + (h0 + hl)) * Hdim + o0 + ol;
    *(unsigned*)(wth + base)     = ph0.u;
    *(unsigned*)(wth + base + 2) = ph1.u;
    *(unsigned*)(wtl + base)     = pl0.u;
    *(unsigned*)(wtl + base + 2) = pl1.u;
  }
}

// ---------------------------------------------------------------------------
// 4) grouped expert GEMM from pre-split bf16 planes.
//    Block tile M=64 x N=128, K-chunk 32; 8 waves, 32x32 per wave =
//    12 v_wmma_f32_16x16x32_bf16 per chunk per wave. Staging is pure 16B
//    copies: GLOBAL_LOAD_ASYNC_TO_LDS_B128 with double buffer + one barrier
//    per chunk (s_wait_asynccnt), falling back to vector copies if the
//    builtin is unavailable.
// ---------------------------------------------------------------------------
constexpr int TM = 64, TN = 128, TK = 32;
constexpr int APAD = 8, BPAD = 8;

__global__ __launch_bounds__(256) void moe_gemm_bf16_kernel(
    const unsigned short* __restrict__ xh, const unsigned short* __restrict__ xl,
    const unsigned short* __restrict__ wth, const unsigned short* __restrict__ wtl,
    const float* __restrict__ bias, const float* __restrict__ gatew,
    const int* __restrict__ perm, const int* __restrict__ counts,
    const int* __restrict__ offsets, float* __restrict__ out) {
  const int e    = blockIdx.y;
  const int cnt  = counts[e];
  const int row0 = blockIdx.z * TM;
  if (row0 >= cnt) return;
  const int n0  = blockIdx.x * TN;
  const int off = offsets[e];

  __shared__ __bf16 Ah[2][TM][TK + APAD];
  __shared__ __bf16 Al[2][TM][TK + APAD];
  __shared__ __bf16 Bh[2][TK][TN + BPAD];
  __shared__ __bf16 Bl[2][TK][TN + BPAD];
  __shared__ int   tokLds[TM];
  __shared__ float gLds[TM];

  const int tid = threadIdx.x;
  if (tid < TM) {
    const int r = row0 + tid;
    if (r < cnt) {
      const int t = perm[off + r];
      tokLds[tid] = t;
      gLds[tid]   = gatew[t];
    } else {
      tokLds[tid] = -1;
      gLds[tid]   = 0.0f;
    }
  }
  __syncthreads();

  const int lane = tid & 31;
  const int wv   = tid >> 5;
  const int wmq  = wv & 1;
  const int wnq  = wv >> 1;

  // copy coordinates (16B per op). A: 4 thr/row, B: 8 thr/row.
  const int arow = tid >> 2;                 // 0..63
  const int acol = (tid & 3) * 8;            // 0,8,16,24 (elements)
  const int atokRaw = tokLds[arow];
  const int atok = atokRaw < 0 ? 0 : atokRaw;     // rows >= cnt: junk is never stored
  const size_t aBase = (size_t)atok * Hdim + acol;
  const int brow = tid >> 3;                 // 0..31 (k)
  const int bcol = (tid & 7) * 16;           // 0..112 (elements)
  const size_t bBase = (size_t)e * Hdim * Hdim + (size_t)brow * Hdim + n0 + bcol;

  auto issueCopies = [&](int buf, int k0) {
#if HAVE_ASYNC_LDS
    async_cp16(xh + aBase + k0, &Ah[buf][arow][acol]);
    async_cp16(xl + aBase + k0, &Al[buf][arow][acol]);
    const size_t bk = bBase + (size_t)k0 * Hdim;
    async_cp16(wth + bk,     &Bh[buf][brow][bcol]);
    async_cp16(wth + bk + 8, &Bh[buf][brow][bcol + 8]);
    async_cp16(wtl + bk,     &Bl[buf][brow][bcol]);
    async_cp16(wtl + bk + 8, &Bl[buf][brow][bcol + 8]);
#else
    const uint4 a0 = *(const uint4*)(xh + aBase + k0);
    const uint4 a1 = *(const uint4*)(xl + aBase + k0);
    const size_t bk = bBase + (size_t)k0 * Hdim;
    const uint4 b0 = *(const uint4*)(wth + bk);
    const uint4 b1 = *(const uint4*)(wth + bk + 8);
    const uint4 b2 = *(const uint4*)(wtl + bk);
    const uint4 b3 = *(const uint4*)(wtl + bk + 8);
    *(uint4*)&Ah[buf][arow][acol]     = a0;
    *(uint4*)&Al[buf][arow][acol]     = a1;
    *(uint4*)&Bh[buf][brow][bcol]     = b0;
    *(uint4*)&Bh[buf][brow][bcol + 8] = b1;
    *(uint4*)&Bl[buf][brow][bcol]     = b2;
    *(uint4*)&Bl[buf][brow][bcol + 8] = b3;
#endif
  };

  FragC c[2][2];
#pragma unroll
  for (int m = 0; m < 2; ++m)
#pragma unroll
    for (int n = 0; n < 2; ++n) c[m][n].v = (f32x8){};

  // fragment read coordinates (per ISA 16-bit layouts)
  const int amBase = wmq * 32 + (lane & 15);
  const int akb    = (lane >> 4) * 8;
  const int bnb    = wnq * 32;

  auto computeChunk = [&](int buf) {
    FragAB ah[2], al[2], bh[2], bl[2];
#pragma unroll
    for (int m = 0; m < 2; ++m) {
      const int am = amBase + m * 16;
      ah[m].h[0] = *(const bf16x8*)&Ah[buf][am][akb];
      ah[m].h[1] = *(const bf16x8*)&Ah[buf][am][akb + 16];
      al[m].h[0] = *(const bf16x8*)&Al[buf][am][akb];
      al[m].h[1] = *(const bf16x8*)&Al[buf][am][akb + 16];
    }
#pragma unroll
    for (int n = 0; n < 2; ++n) {
      const int nb = bnb + n * 16;
      for (int s = 0; s < 8; ++s) {
        bh[n].v[s]     = Bh[buf][akb + s][nb + (lane & 15)];
        bh[n].v[s + 8] = Bh[buf][akb + 16 + s][nb + (lane & 15)];
        bl[n].v[s]     = Bl[buf][akb + s][nb + (lane & 15)];
        bl[n].v[s + 8] = Bl[buf][akb + 16 + s][nb + (lane & 15)];
      }
    }
#pragma unroll
    for (int m = 0; m < 2; ++m) {
#pragma unroll
      for (int n = 0; n < 2; ++n) {
        c[m][n].v = __builtin_amdgcn_wmma_f32_16x16x32_bf16(
            false, ah[m].v, false, bh[n].v, (short)0, c[m][n].v, false, false);
        c[m][n].v = __builtin_amdgcn_wmma_f32_16x16x32_bf16(
            false, ah[m].v, false, bl[n].v, (short)0, c[m][n].v, false, false);
        c[m][n].v = __builtin_amdgcn_wmma_f32_16x16x32_bf16(
            false, al[m].v, false, bh[n].v, (short)0, c[m][n].v, false, false);
      }
    }
  };

#if HAVE_ASYNC_LDS
  // double buffer, one barrier per chunk:
  // [wait asynccnt 0][barrier][issue buf^1][compute buf]
  issueCopies(0, 0);
  int buf = 0;
  for (int k0 = 0; k0 < Hdim; k0 += TK) {
    wait_async0();
    __syncthreads();
    if (k0 + TK < Hdim) issueCopies(buf ^ 1, k0 + TK);
    computeChunk(buf);
    buf ^= 1;
  }
#else
  for (int k0 = 0; k0 < Hdim; k0 += TK) {
    issueCopies(0, k0);
    __syncthreads();
    computeChunk(0);
    __syncthreads();
  }
#endif

  // ---- epilogue: out[tok][o] = gate[tok] * (acc + bias[e][o]) ----
  const int nl   = lane & 15;
  const int half = lane >> 4;
#pragma unroll
  for (int n = 0; n < 2; ++n) {
    const int   o   = n0 + wnq * 32 + n * 16 + nl;
    const float bvv = bias[(size_t)e * Hdim + o];
#pragma unroll
    for (int m = 0; m < 2; ++m) {
#pragma unroll
      for (int r = 0; r < 8; ++r) {
        const int ml   = wmq * 32 + m * 16 + r + half * 8;
        const int grow = row0 + ml;
        if (grow < cnt) {
          out[(size_t)tokLds[ml] * Hdim + o] = gLds[ml] * (c[m][n].f[r] + bvv);
        }
      }
    }
  }
}

// ---------------------------------------------------------------------------
// 4-fallback) f32-staging GEMM (round-2 version) for small workspaces
// ---------------------------------------------------------------------------
__global__ __launch_bounds__(256) void moe_gemm_f32_kernel(
    const float* __restrict__ x, const float* __restrict__ W,
    const float* __restrict__ bias, const float* __restrict__ gatew,
    const int* __restrict__ perm, const int* __restrict__ counts,
    const int* __restrict__ offsets, float* __restrict__ out) {
  const int e    = blockIdx.y;
  const int cnt  = counts[e];
  const int row0 = blockIdx.z * TM;
  if (row0 >= cnt) return;
  const int n0  = blockIdx.x * TN;
  const int off = offsets[e];

  __shared__ __bf16 Ah[TM][TK + APAD];
  __shared__ __bf16 Al[TM][TK + APAD];
  __shared__ __bf16 Bh[TK][TN + BPAD];
  __shared__ __bf16 Bl[TK][TN + BPAD];
  __shared__ int   tokLds[TM];
  __shared__ float gLds[TM];

  const int tid = threadIdx.x;
  if (tid < TM) {
    const int r = row0 + tid;
    if (r < cnt) {
      const int t = perm[off + r];
      tokLds[tid] = t;
      gLds[tid]   = gatew[t];
    } else {
      tokLds[tid] = -1;
      gLds[tid]   = 0.0f;
    }
  }
  __syncthreads();

  const int lane = tid & 31;
  const int wv   = tid >> 5;
  const int wmq  = wv & 1;
  const int wnq  = wv >> 1;

  const float* __restrict__ Wb = W + (size_t)e * Hdim * Hdim;
  const int lr = tid >> 3;
  const int lk = (tid & 7) * 4;
  int atok[2];
  atok[0] = tokLds[lr];
  atok[1] = tokLds[lr + 32];

  float4 aReg[2], bReg[4];
  auto loadA = [&](int k0) {
#pragma unroll
    for (int p = 0; p < 2; ++p) {
      aReg[p] = make_float4(0.f, 0.f, 0.f, 0.f);
      if (atok[p] >= 0)
        aReg[p] = *(const float4*)(x + (size_t)atok[p] * Hdim + k0 + lk);
    }
  };
  auto loadB = [&](int k0) {
#pragma unroll
    for (int p = 0; p < 4; ++p)
      bReg[p] = *(const float4*)(Wb + (size_t)(n0 + lr + p * 32) * Hdim + k0 + lk);
  };
  loadA(0);
  loadB(0);

  FragC c[2][2];
#pragma unroll
  for (int m = 0; m < 2; ++m)
#pragma unroll
    for (int n = 0; n < 2; ++n) c[m][n].v = (f32x8){};

  const int amBase = wmq * 32 + (lane & 15);
  const int akb    = (lane >> 4) * 8;
  const int bnb    = wnq * 32;

  for (int k0 = 0; k0 < Hdim; k0 += TK) {
#pragma unroll
    for (int p = 0; p < 2; ++p) {
      const int  row  = lr + p * 32;
      const float v[4] = {aReg[p].x, aReg[p].y, aReg[p].z, aReg[p].w};
#pragma unroll
      for (int j = 0; j < 4; j += 2) {
        Pack2 ph, pl;
        ph.b[0] = (__bf16)v[j];
        ph.b[1] = (__bf16)v[j + 1];
        pl.b[0] = (__bf16)(v[j]     - (float)ph.b[0]);
        pl.b[1] = (__bf16)(v[j + 1] - (float)ph.b[1]);
        *(unsigned*)&Ah[row][lk + j] = ph.u;
        *(unsigned*)&Al[row][lk + j] = pl.u;
      }
    }
#pragma unroll
    for (int p = 0; p < 4; ++p) {
      const int  o    = lr + p * 32;
      const float v[4] = {bReg[p].x, bReg[p].y, bReg[p].z, bReg[p].w};
#pragma unroll
      for (int i = 0; i < 4; ++i) {
        const __bf16 h = (__bf16)v[i];
        Bh[lk + i][o] = h;
        Bl[lk + i][o] = (__bf16)(v[i] - (float)h);
      }
    }
    __syncthreads();

    if (k0 + TK < Hdim) { loadA(k0 + TK); loadB(k0 + TK); }

    FragAB ah[2], al[2], bh[2], bl[2];
#pragma unroll
    for (int m = 0; m < 2; ++m) {
      const int am = amBase + m * 16;
      ah[m].h[0] = *(const bf16x8*)&Ah[am][akb];
      ah[m].h[1] = *(const bf16x8*)&Ah[am][akb + 16];
      al[m].h[0] = *(const bf16x8*)&Al[am][akb];
      al[m].h[1] = *(const bf16x8*)&Al[am][akb + 16];
    }
#pragma unroll
    for (int n = 0; n < 2; ++n) {
      const int nb = bnb + n * 16;
      for (int s = 0; s < 8; ++s) {
        bh[n].v[s]     = Bh[akb + s][nb + (lane & 15)];
        bh[n].v[s + 8] = Bh[akb + 16 + s][nb + (lane & 15)];
        bl[n].v[s]     = Bl[akb + s][nb + (lane & 15)];
        bl[n].v[s + 8] = Bl[akb + 16 + s][nb + (lane & 15)];
      }
    }
#pragma unroll
    for (int m = 0; m < 2; ++m) {
#pragma unroll
      for (int n = 0; n < 2; ++n) {
        c[m][n].v = __builtin_amdgcn_wmma_f32_16x16x32_bf16(
            false, ah[m].v, false, bh[n].v, (short)0, c[m][n].v, false, false);
        c[m][n].v = __builtin_amdgcn_wmma_f32_16x16x32_bf16(
            false, ah[m].v, false, bl[n].v, (short)0, c[m][n].v, false, false);
        c[m][n].v = __builtin_amdgcn_wmma_f32_16x16x32_bf16(
            false, al[m].v, false, bh[n].v, (short)0, c[m][n].v, false, false);
      }
    }
    __syncthreads();
  }

  const int nl   = lane & 15;
  const int half = lane >> 4;
#pragma unroll
  for (int n = 0; n < 2; ++n) {
    const int   o   = n0 + wnq * 32 + n * 16 + nl;
    const float bvv = bias[(size_t)e * Hdim + o];
#pragma unroll
    for (int m = 0; m < 2; ++m) {
#pragma unroll
      for (int r = 0; r < 8; ++r) {
        const int ml   = wmq * 32 + m * 16 + r + half * 8;
        const int grow = row0 + ml;
        if (grow < cnt) {
          out[(size_t)tokLds[ml] * Hdim + o] = gLds[ml] * (c[m][n].f[r] + bvv);
        }
      }
    }
  }
}

// ---------------------------------------------------------------------------
// launch
// ---------------------------------------------------------------------------
extern "C" void kernel_launch(void* const* d_in, const int* in_sizes, int n_in,
                              void* d_out, int out_size, void* d_ws, size_t ws_size,
                              hipStream_t stream) {
  (void)in_sizes; (void)n_in; (void)out_size;
  const float* x  = (const float*)d_in[0];   // [B,S,H]
  const float* wg = (const float*)d_in[1];   // [H,E]
  const float* W  = (const float*)d_in[2];   // [E,H,H] (out, in)
  const float* b  = (const float*)d_in[3];   // [E,H]
  float* out = (float*)d_out;

  char* ws      = (char*)d_ws;
  int*   eid    = (int*)(ws);                        // 8192 ints
  float* gw     = (float*)(ws + 4 * Ntok);           // 8192 floats
  int*   perm   = (int*)(ws + 8 * Ntok);             // 8192 ints
  int*   counts = (int*)(ws + 12 * Ntok);            // 8
  int*   offs   = counts + Edim;                     // 8
  int*   curs   = counts + 2 * Edim;                 // 8

  moe_init_kernel<<<1, 32, 0, stream>>>(counts);
  moe_gate_kernel<<<Ntok / 8, 256, 0, stream>>>(x, wg, eid, gw, counts);
  moe_scan_kernel<<<1, 32, 0, stream>>>(counts, offs, curs);
  moe_scatter_kernel<<<Ntok / 256, 256, 0, stream>>>(eid, offs, curs, perm);

  const size_t metaEnd = 131072;
  const size_t planeX  = (size_t)Ntok * Hdim * 2;          // 32 MiB
  const size_t planeW  = (size_t)Edim * Hdim * Hdim * 2;   // 64 MiB
  const size_t need    = metaEnd + 2 * planeX + 2 * planeW;

  dim3 grid(Hdim / TN, Edim, Ntok / TM);   // (16, 8, 128); empty tiles early-exit

  if (ws_size >= need) {
    unsigned short* xh  = (unsigned short*)(ws + metaEnd);
    unsigned short* xl  = (unsigned short*)(ws + metaEnd + planeX);
    unsigned short* wth = (unsigned short*)(ws + metaEnd + 2 * planeX);
    unsigned short* wtl = (unsigned short*)(ws + metaEnd + 2 * planeX + planeW);

    moe_xconv_kernel<<<(Ntok * (size_t)Hdim) / (256 * 4), 256, 0, stream>>>(x, xh, xl);
    dim3 gw2(Hdim / 32, Hdim / 32, Edim);
    moe_wconv_kernel<<<gw2, 256, 0, stream>>>(W, wth, wtl);
    moe_gemm_bf16_kernel<<<grid, 256, 0, stream>>>(xh, xl, wth, wtl, b, gw,
                                                   perm, counts, offs, out);
  } else {
    moe_gemm_f32_kernel<<<grid, 256, 0, stream>>>(x, W, b, gw, perm, counts, offs, out);
  }
}
